// Repro_15075335209178
// MI455X (gfx1250) — matrix operand order repaired and checked by reading, not compiled, not run on hardware
//
#include <hip/hip_runtime.h>
#include <hip/hip_bf16.h>
#include <math.h>
#include <stdint.h>

typedef __attribute__((ext_vector_type(16))) _Float16 v16h;
typedef __attribute__((ext_vector_type(8)))  _Float16 v8h;
typedef __attribute__((ext_vector_type(8)))  float    v8f;
typedef __attribute__((ext_vector_type(4)))  float    v4f;

#define DEV __device__ __forceinline__

namespace {

constexpr int B_   = 500;
constexpr int S_   = 128;
constexpr int E_   = 128;
constexpr int H_   = 8;
constexpr int HD_  = 16;
constexpr int DIN_ = 512;
constexpr int DFF_ = 512;
constexpr int LDC_ = 132;            // padded LDS row stride (floats)
constexpr float SCALE_ = 1.0f / 0.9f;
constexpr float EPS_   = 1e-5f;
constexpr float INV_E  = 1.0f / 128.0f;

DEV v8f wmma32(v16h a, v16h b, v8f c) {
  return __builtin_amdgcn_wmma_f32_16x16x32_f16(false, a, false, b, (short)0, c,
                                                false, false);
}

DEV v16h cat16(v8h lo, v8h hi) {
  return __builtin_shufflevector(lo, hi, 0, 1, 2, 3, 4, 5, 6, 7, 8, 9, 10, 11,
                                 12, 13, 14, 15);
}

// A fragment 16x32 from row-major f16 (lda in elems). ISA layout:
// lane<16: halves0-7=K0..7, halves8-15=K16..23 ; lane>=16: +8 on K.
DEV v16h load_a_f16(const _Float16* A, int lda, int kbase) {
  int lane = threadIdx.x & 31;
  int row  = lane & 15;
  int koff = (lane & 16) ? 8 : 0;
  const _Float16* p = A + (size_t)row * lda + kbase + koff;
  v8h lo = *(const v8h*)p;
  v8h hi = *(const v8h*)(p + 16);
  return cat16(lo, hi);
}

// A fragment from row-major f32 source (converted to f16 in regs).
DEV v16h load_a_f32(const float* A, int lda, int kbase) {
  int lane = threadIdx.x & 31;
  int row  = lane & 15;
  int koff = (lane & 16) ? 8 : 0;
  const float* p = A + (size_t)row * lda + kbase + koff;
  v4f a0 = *(const v4f*)p;
  v4f a1 = *(const v4f*)(p + 4);
  v4f b0 = *(const v4f*)(p + 16);
  v4f b1 = *(const v4f*)(p + 20);
  v16h a;
#pragma unroll
  for (int j = 0; j < 4; ++j) {
    a[j]      = (_Float16)a0[j];
    a[j + 4]  = (_Float16)a1[j];
    a[j + 8]  = (_Float16)b0[j];
    a[j + 12] = (_Float16)b1[j];
  }
  return a;
}

// B fragment 32x16 where W is [N][K] row-major:
// lane<16: col N=n0+lane, halves0-7=K0..7, halves8-15=K16..23 ; lane>=16: +8 on K.
DEV v16h load_b_f16(const _Float16* W, int ldk, int n0, int kbase) {
  int lane = threadIdx.x & 31;
  int n    = lane & 15;
  int kb   = kbase + ((lane & 16) ? 8 : 0);
  const _Float16* p = W + (size_t)(n0 + n) * ldk + kb;
  v8h lo = *(const v8h*)p;
  v8h hi = *(const v8h*)(p + 16);
  return cat16(lo, hi);
}

DEV float rowred_sum(float v) {
#pragma unroll
  for (int off = 1; off < 16; off <<= 1) v += __shfl_xor(v, off, 32);
  return v;
}
DEV float rowred_max(float v) {
#pragma unroll
  for (int off = 1; off < 16; off <<= 1) v = fmaxf(v, __shfl_xor(v, off, 32));
  return v;
}

// ---------------- weight conversion ----------------
__global__ void k_cvt(const float* __restrict__ src, _Float16* __restrict__ dst,
                      int n) {
  int i = blockIdx.x * 256 + threadIdx.x;
  if (i < n) dst[i] = (_Float16)src[i];
}

// ---------------- 1) input projection + dropout + pos + LN1 ----------------
__global__ __launch_bounds__(256) void k_inproj(
    const float* __restrict__ x, const _Float16* __restrict__ w_in_h,
    const float* __restrict__ b_in, const float* __restrict__ pos,
    const float* __restrict__ ln1w, const float* __restrict__ ln1b,
    const unsigned char* __restrict__ m1, float* __restrict__ res1,
    _Float16* __restrict__ y1) {
  __shared__ float lds_c[16 * LDC_];
  int m0   = blockIdx.x * 16;  // flat row = b*S + s
  int wave = threadIdx.x >> 5;
  int n0   = wave * 16;
  const float* xa = x + (size_t)m0 * DIN_;
  v8f c = {};
#pragma unroll 4
  for (int kb = 0; kb < DIN_; kb += 32) {
    v16h a = load_a_f32(xa, DIN_, kb);
    v16h b = load_b_f16(w_in_h, DIN_, n0, kb);
    c = wmma32(a, b, c);
  }
  int lane = threadIdx.x & 31;
  int n = lane & 15, mb = (lane & 16) ? 8 : 0;
#pragma unroll
  for (int r = 0; r < 8; ++r) lds_c[(mb + r) * LDC_ + n0 + n] = c[r];
  __syncthreads();

  int sub  = threadIdx.x & 15;      // 8-col chunk within row
  int lrow = threadIdx.x >> 4;      // 0..15
  size_t rowg = (size_t)m0 + lrow;
  int s = (int)(rowg & (S_ - 1));
  int j0 = sub * 8;
  uint64_t mb8;
  __builtin_memcpy(&mb8, m1 + rowg * E_ + j0, 8);
  float vv[8];
  float sum = 0.f, sq = 0.f;
#pragma unroll
  for (int j = 0; j < 8; ++j) {
    float v = lds_c[lrow * LDC_ + j0 + j] + b_in[j0 + j];
    v = ((mb8 >> (8 * j)) & 0xff) ? v * SCALE_ : 0.f;
    v += pos[s * E_ + j0 + j];
    vv[j] = v;
    sum += v; sq += v * v;
  }
  sum = rowred_sum(sum);
  sq  = rowred_sum(sq);
  float mu = sum * INV_E;
  float rstd = rsqrtf(sq * INV_E - mu * mu + EPS_);
  v4f r0, r1;
  v8h yv;
#pragma unroll
  for (int j = 0; j < 4; ++j) {
    r0[j] = vv[j];
    r1[j] = vv[j + 4];
  }
#pragma unroll
  for (int j = 0; j < 8; ++j)
    yv[j] = (_Float16)((vv[j] - mu) * rstd * ln1w[j0 + j] + ln1b[j0 + j]);
  *(v4f*)(res1 + rowg * E_ + j0)     = r0;
  *(v4f*)(res1 + rowg * E_ + j0 + 4) = r1;
  *(v8h*)(y1 + rowg * E_ + j0)       = yv;
}

// ---------------- 2) QKV projection + head split ----------------
__global__ __launch_bounds__(256) void k_qkv(
    const _Float16* __restrict__ y1, const _Float16* __restrict__ wqkv_h,
    const float* __restrict__ b_qkv, _Float16* __restrict__ qf,
    _Float16* __restrict__ kf, _Float16* __restrict__ vT) {
  int m0   = blockIdx.x * 16;
  int wave = threadIdx.x >> 5;
  int f0   = blockIdx.y * 128 + wave * 16;
  const _Float16* ya = y1 + (size_t)m0 * E_;
  v8f c = {};
#pragma unroll
  for (int kb = 0; kb < E_; kb += 32) {
    v16h a = load_a_f16(ya, E_, kb);
    v16h b = load_b_f16(wqkv_h, E_, f0, kb);
    c = wmma32(a, b, c);
  }
  int lane = threadIdx.x & 31;
  int n = lane & 15, mb = (lane & 16) ? 8 : 0;
  int f   = f0 + n;
  int idx = f >> 7, hh = (f >> 4) & 7, d = f & 15;
  float bias = b_qkv[f];
#pragma unroll
  for (int r = 0; r < 8; ++r) {
    int rowg = m0 + mb + r;
    int bb = rowg >> 7, s = rowg & (S_ - 1);
    float val = c[r] + bias;
    size_t base = (size_t)(bb * H_ + hh);
    if (idx == 0)      qf[(base * S_ + s) * HD_ + d] = (_Float16)(val * 0.25f);
    else if (idx == 1) kf[(base * S_ + s) * HD_ + d] = (_Float16)val;
    else               vT[(base * HD_ + d) * S_ + s] = (_Float16)val;
  }
}

// ---------------- 3) attention (per b,h; S resident) ----------------
__global__ __launch_bounds__(256) void k_attn(
    const _Float16* __restrict__ qf, const _Float16* __restrict__ kf,
    const _Float16* __restrict__ vT, const float* __restrict__ attn_bias,
    const float* __restrict__ key_pad, const unsigned char* __restrict__ m_attn,
    _Float16* __restrict__ obuf) {
  __shared__ _Float16 probs[8][16 * 128];
  int bh = blockIdx.x;
  int b = bh >> 3, h = bh & 7;
  int wave = threadIdx.x >> 5;
  int lane = threadIdx.x & 31;
  int l0  = wave * 16;
  int row = lane & 15;
  int mb  = (lane & 16) ? 8 : 0;
  const _Float16* qb  = qf + (size_t)bh * S_ * HD_;
  const _Float16* kb_ = kf + (size_t)bh * S_ * HD_;
  const _Float16* vTb = vT + (size_t)bh * HD_ * S_;
  const float* biasb  = attn_bias + (size_t)bh * S_ * S_;
  const unsigned char* maskb = m_attn + (size_t)bh * S_ * S_;

  // A = q tile (K=16, zero-padded to 32)
  v16h a;
  {
    int koff = (lane & 16) ? 8 : 0;
    v8h lo = *(const v8h*)(qb + (l0 + row) * HD_ + koff);
    v8h z = {};
    a = cat16(lo, z);
  }
  float cc[8][8];
#pragma unroll
  for (int t = 0; t < 8; ++t) {
    v16h bf;
    {
      const _Float16* p = kb_ + (t * 16 + row) * HD_ + ((lane & 16) ? 8 : 0);
      v8h zb = {};
      bf = cat16(*(const v8h*)p, zb);
    }
    v8f c0 = {};
    c0 = wmma32(a, bf, c0);
#pragma unroll
    for (int r = 0; r < 8; ++r) cc[t][r] = c0[r];
  }
#pragma unroll
  for (int t = 0; t < 8; ++t)
#pragma unroll
    for (int r = 0; r < 8; ++r) {
      int m = l0 + mb + r, sx = t * 16 + row;
      cc[t][r] += biasb[(size_t)m * S_ + sx] + key_pad[b * S_ + sx];
    }
#pragma unroll
  for (int r = 0; r < 8; ++r) {
    float mx = -3.0e38f;
#pragma unroll
    for (int t = 0; t < 8; ++t) mx = fmaxf(mx, cc[t][r]);
    mx = rowred_max(mx);
    float sum = 0.f;
#pragma unroll
    for (int t = 0; t < 8; ++t) {
      float e = __expf(cc[t][r] - mx);
      cc[t][r] = e;
      sum += e;
    }
    sum = rowred_sum(sum);
    float inv = 1.0f / sum;
    int m = l0 + mb + r;
#pragma unroll
    for (int t = 0; t < 8; ++t) {
      int sx = t * 16 + row;
      float p = cc[t][r] * inv;
      p = maskb[(size_t)m * S_ + sx] ? p * SCALE_ : 0.f;
      probs[wave][(mb + r) * 128 + sx] = (_Float16)p;
    }
  }
  __syncthreads();
  // O = probs @ v  (K = S = 128)
  v8f oc = {};
#pragma unroll
  for (int kb2 = 0; kb2 < S_; kb2 += 32) {
    v16h pa = load_a_f16(&probs[wave][0], 128, kb2);
    v16h vb = load_b_f16(vTb, S_, 0, kb2);
    oc = wmma32(pa, vb, oc);
  }
#pragma unroll
  for (int r = 0; r < 8; ++r)
    obuf[((size_t)b * S_ + (l0 + mb + r)) * E_ + h * HD_ + row] = (_Float16)oc[r];
}

// ---------------- 4) out projection + LN2 + drop + residual + LN3 ----------------
__global__ __launch_bounds__(256) void k_outproj(
    const _Float16* __restrict__ obuf, const _Float16* __restrict__ wout_h,
    const float* __restrict__ b_out, const float* __restrict__ ln2w,
    const float* __restrict__ ln2b, const float* __restrict__ ln3w,
    const float* __restrict__ ln3b, const unsigned char* __restrict__ m2,
    const float* __restrict__ res1, float* __restrict__ res2,
    _Float16* __restrict__ y3) {
  __shared__ float lds_c[16 * LDC_];
  int m0 = blockIdx.x * 16;
  int wave = threadIdx.x >> 5;
  int n0 = wave * 16;
  const _Float16* oa = obuf + (size_t)m0 * E_;
  v8f c = {};
#pragma unroll
  for (int kb = 0; kb < E_; kb += 32) {
    v16h a = load_a_f16(oa, E_, kb);
    v16h b = load_b_f16(wout_h, E_, n0, kb);
    c = wmma32(a, b, c);
  }
  int lane = threadIdx.x & 31;
  int n = lane & 15, mb = (lane & 16) ? 8 : 0;
#pragma unroll
  for (int r = 0; r < 8; ++r) lds_c[(mb + r) * LDC_ + n0 + n] = c[r];
  __syncthreads();

  int sub = threadIdx.x & 15;
  int lrow = threadIdx.x >> 4;
  size_t rowg = (size_t)m0 + lrow;
  int j0 = sub * 8;
  float vv[8];
  float sum = 0.f, sq = 0.f;
#pragma unroll
  for (int j = 0; j < 8; ++j) {
    float v = lds_c[lrow * LDC_ + j0 + j] + b_out[j0 + j];
    vv[j] = v;
    sum += v; sq += v * v;
  }
  sum = rowred_sum(sum);
  sq  = rowred_sum(sq);
  float mu = sum * INV_E;
  float rstd = rsqrtf(sq * INV_E - mu * mu + EPS_);
  uint64_t mb8;
  __builtin_memcpy(&mb8, m2 + rowg * E_ + j0, 8);
  v4f p0 = *(const v4f*)(res1 + rowg * E_ + j0);
  v4f p1 = *(const v4f*)(res1 + rowg * E_ + j0 + 4);
  float sum2 = 0.f, sq2 = 0.f;
#pragma unroll
  for (int j = 0; j < 8; ++j) {
    float v = (vv[j] - mu) * rstd * ln2w[j0 + j] + ln2b[j0 + j];
    v = ((mb8 >> (8 * j)) & 0xff) ? v * SCALE_ : 0.f;
    float r2 = v + (j < 4 ? p0[j] : p1[j - 4]);
    vv[j] = r2;
    sum2 += r2; sq2 += r2 * r2;
  }
  sum2 = rowred_sum(sum2);
  sq2  = rowred_sum(sq2);
  float mu2 = sum2 * INV_E;
  float rstd2 = rsqrtf(sq2 * INV_E - mu2 * mu2 + EPS_);
  v4f r0, r1;
  v8h yv;
#pragma unroll
  for (int j = 0; j < 4; ++j) {
    r0[j] = vv[j];
    r1[j] = vv[j + 4];
  }
#pragma unroll
  for (int j = 0; j < 8; ++j)
    yv[j] = (_Float16)((vv[j] - mu2) * rstd2 * ln3w[j0 + j] + ln3b[j0 + j]);
  *(v4f*)(res2 + rowg * E_ + j0)     = r0;
  *(v4f*)(res2 + rowg * E_ + j0 + 4) = r1;
  *(v8h*)(y3 + rowg * E_ + j0)       = yv;
}

// ---------------- 5) FF1 + exact GELU + dropout ----------------
__global__ __launch_bounds__(256) void k_ff1(
    const _Float16* __restrict__ y3, const _Float16* __restrict__ wff1_h,
    const float* __restrict__ b_ff1, const unsigned char* __restrict__ m3,
    _Float16* __restrict__ ff) {
  int m0 = blockIdx.x * 16;
  int wave = threadIdx.x >> 5;
  int f0 = blockIdx.y * 128 + wave * 16;
  const _Float16* ya = y3 + (size_t)m0 * E_;
  v8f c = {};
#pragma unroll
  for (int kb = 0; kb < E_; kb += 32) {
    v16h a = load_a_f16(ya, E_, kb);
    v16h b = load_b_f16(wff1_h, E_, f0, kb);
    c = wmma32(a, b, c);
  }
  int lane = threadIdx.x & 31;
  int n = lane & 15, mb = (lane & 16) ? 8 : 0;
  int f = f0 + n;
  float bias = b_ff1[f];
#pragma unroll
  for (int r = 0; r < 8; ++r) {
    size_t rowg = (size_t)m0 + mb + r;
    float v = c[r] + bias;
    float g = 0.5f * v * (1.0f + erff(v * 0.70710678f));  // exact erf GELU
    g = m3[rowg * DFF_ + f] ? g * SCALE_ : 0.f;
    ff[rowg * DFF_ + f] = (_Float16)g;
  }
}

// ---------------- 6) FF2 + dropout + residual + LN4 -> out ----------------
__global__ __launch_bounds__(256) void k_ff2(
    const _Float16* __restrict__ ff, const _Float16* __restrict__ wff2_h,
    const float* __restrict__ b_ff2, const float* __restrict__ ln4w,
    const float* __restrict__ ln4b, const unsigned char* __restrict__ m4,
    const float* __restrict__ res2, float* __restrict__ out) {
  __shared__ float lds_c[16 * LDC_];
  int m0 = blockIdx.x * 16;
  int wave = threadIdx.x >> 5;
  int n0 = wave * 16;
  const _Float16* fa = ff + (size_t)m0 * DFF_;
  v8f c = {};
#pragma unroll 4
  for (int kb = 0; kb < DFF_; kb += 32) {
    v16h a = load_a_f16(fa, DFF_, kb);
    v16h b = load_b_f16(wff2_h, DFF_, n0, kb);
    c = wmma32(a, b, c);
  }
  int lane = threadIdx.x & 31;
  int n = lane & 15, mb = (lane & 16) ? 8 : 0;
#pragma unroll
  for (int r = 0; r < 8; ++r) lds_c[(mb + r) * LDC_ + n0 + n] = c[r];
  __syncthreads();

  int sub = threadIdx.x & 15;
  int lrow = threadIdx.x >> 4;
  size_t rowg = (size_t)m0 + lrow;
  int j0 = sub * 8;
  uint64_t mb8;
  __builtin_memcpy(&mb8, m4 + rowg * E_ + j0, 8);
  v4f p0 = *(const v4f*)(res2 + rowg * E_ + j0);
  v4f p1 = *(const v4f*)(res2 + rowg * E_ + j0 + 4);
  float vv[8];
  float sum = 0.f, sq = 0.f;
#pragma unroll
  for (int j = 0; j < 8; ++j) {
    float v = lds_c[lrow * LDC_ + j0 + j] + b_ff2[j0 + j];
    v = ((mb8 >> (8 * j)) & 0xff) ? v * SCALE_ : 0.f;
    v += (j < 4 ? p0[j] : p1[j - 4]);
    vv[j] = v;
    sum += v; sq += v * v;
  }
  sum = rowred_sum(sum);
  sq  = rowred_sum(sq);
  float mu = sum * INV_E;
  float rstd = rsqrtf(sq * INV_E - mu * mu + EPS_);
  v4f o0, o1;
#pragma unroll
  for (int j = 0; j < 4; ++j) {
    o0[j] = (vv[j] - mu) * rstd * ln4w[j0 + j] + ln4b[j0 + j];
    o1[j] = (vv[j + 4] - mu) * rstd * ln4w[j0 + j + 4] + ln4b[j0 + j + 4];
  }
  *(v4f*)(out + rowg * E_ + j0)     = o0;
  *(v4f*)(out + rowg * E_ + j0 + 4) = o1;
}

}  // namespace

extern "C" void kernel_launch(void* const* d_in, const int* in_sizes, int n_in,
                              void* d_out, int out_size, void* d_ws,
                              size_t ws_size, hipStream_t stream) {
  (void)in_sizes; (void)n_in; (void)out_size; (void)ws_size;
  const float* x         = (const float*)d_in[0];
  const float* w_in      = (const float*)d_in[1];
  const float* b_in      = (const float*)d_in[2];
  const float* pos       = (const float*)d_in[3];
  const float* ln1w      = (const float*)d_in[4];
  const float* ln1b      = (const float*)d_in[5];
  const float* w_qkv     = (const float*)d_in[6];
  const float* b_qkv     = (const float*)d_in[7];
  const float* w_out     = (const float*)d_in[8];
  const float* b_out     = (const float*)d_in[9];
  const float* key_pad   = (const float*)d_in[10];
  const float* attn_bias = (const float*)d_in[11];
  const float* ln2w      = (const float*)d_in[12];
  const float* ln2b      = (const float*)d_in[13];
  const float* ln3w      = (const float*)d_in[14];
  const float* ln3b      = (const float*)d_in[15];
  const float* w_ff1     = (const float*)d_in[16];
  const float* b_ff1     = (const float*)d_in[17];
  const float* w_ff2     = (const float*)d_in[18];
  const float* b_ff2     = (const float*)d_in[19];
  const float* ln4w      = (const float*)d_in[20];
  const float* ln4b      = (const float*)d_in[21];
  const unsigned char* m1     = (const unsigned char*)d_in[22];
  const unsigned char* m_attn = (const unsigned char*)d_in[23];
  const unsigned char* m2     = (const unsigned char*)d_in[24];
  const unsigned char* m3     = (const unsigned char*)d_in[25];
  const unsigned char* m4     = (const unsigned char*)d_in[26];
  float* out = (float*)d_out;

  char* wp = (char*)d_ws;
  auto take = [&](size_t bytes) -> char* {
    char* r = wp;
    wp += (bytes + 255) & ~(size_t)255;
    return r;
  };
  const size_t BSE = (size_t)B_ * S_ * E_;
  _Float16* w_in_h  = (_Float16*)take((size_t)E_ * DIN_ * 2);
  _Float16* w_qkv_h = (_Float16*)take((size_t)3 * E_ * E_ * 2);
  _Float16* w_out_h = (_Float16*)take((size_t)E_ * E_ * 2);
  _Float16* w_ff1_h = (_Float16*)take((size_t)DFF_ * E_ * 2);
  _Float16* w_ff2_h = (_Float16*)take((size_t)E_ * DFF_ * 2);
  float*    res1 = (float*)take(BSE * 4);
  _Float16* y1   = (_Float16*)take(BSE * 2);
  _Float16* qf   = (_Float16*)take(BSE * 2);
  _Float16* kf   = (_Float16*)take(BSE * 2);
  _Float16* vT   = (_Float16*)take(BSE * 2);
  _Float16* obuf = (_Float16*)take(BSE * 2);
  float*    res2 = (float*)take(BSE * 4);
  _Float16* y3   = (_Float16*)take(BSE * 2);
  _Float16* ffb  = (_Float16*)take((size_t)B_ * S_ * DFF_ * 2);

  auto cvt = [&](const float* s, _Float16* d, int n) {
    k_cvt<<<(n + 255) / 256, 256, 0, stream>>>(s, d, n);
  };
  cvt(w_in,  w_in_h,  E_ * DIN_);
  cvt(w_qkv, w_qkv_h, 3 * E_ * E_);
  cvt(w_out, w_out_h, E_ * E_);
  cvt(w_ff1, w_ff1_h, DFF_ * E_);
  cvt(w_ff2, w_ff2_h, E_ * DFF_);

  int rowBlocks = (B_ * S_) / 16;  // 4000
  k_inproj<<<rowBlocks, 256, 0, stream>>>(x, w_in_h, b_in, pos, ln1w, ln1b, m1,
                                          res1, y1);
  k_qkv<<<dim3(rowBlocks, 3), 256, 0, stream>>>(y1, w_qkv_h, b_qkv, qf, kf, vT);
  k_attn<<<B_ * H_, 256, 0, stream>>>(qf, kf, vT, attn_bias, key_pad, m_attn,
                                      obuf);
  k_outproj<<<rowBlocks, 256, 0, stream>>>(obuf, w_out_h, b_out, ln2w, ln2b,
                                           ln3w, ln3b, m2, res1, res2, y3);
  k_ff1<<<dim3(rowBlocks, DFF_ / 128), 256, 0, stream>>>(y3, w_ff1_h, b_ff1, m3,
                                                         ffb);
  k_ff2<<<rowBlocks, 256, 0, stream>>>(ffb, w_ff2_h, b_ff2, ln4w, ln4b, m4,
                                       res2, out);
}
